// DADecoderLayer_89309549953426
// MI455X (gfx1250) — matrix operand order repaired and checked by reading, compiled, not run on hardware
//
#include <hip/hip_runtime.h>

// ---------------------------------------------------------------- types
typedef _Float16 v16h __attribute__((ext_vector_type(16)));
typedef _Float16 v8h  __attribute__((ext_vector_type(8)));
typedef float    v8f  __attribute__((ext_vector_type(8)));

union Frag16 { v16h v; v8h h[2]; _Float16 e[16]; };

// ---------------------------------------------------------------- problem constants
constexpr int Bc   = 8;
constexpr int Qc   = 1024;
constexpr int Ec   = 256;
constexpr int NHc  = 8;
constexpr int DHc  = 32;
constexpr int STOT = 21760;   // 128*128 + 64*64 + 32*32 + 16*16

__constant__ int d_HL[4] = {128, 64, 32, 16};           // levels are square
__constant__ int d_ST[4] = {0, 16384, 20480, 21504};

// ---------------------------------------------------------------- DPP row reductions
// Reduce over a 16-lane row (wave32 = two independent rows) using VALU-only
// DPP row rotations: no LDS traffic, no dscnt waits, co-executes with WMMA.
template<int CTRL>
__device__ __forceinline__ float dpp_rot(float v) {
  return __int_as_float(__builtin_amdgcn_update_dpp(
      0, __float_as_int(v), CTRL, 0xf, 0xf, true));
}
__device__ __forceinline__ float rowmax16(float v) {
  v = fmaxf(v, dpp_rot<0x128>(v));   // row_ror:8
  v = fmaxf(v, dpp_rot<0x124>(v));   // row_ror:4
  v = fmaxf(v, dpp_rot<0x122>(v));   // row_ror:2
  v = fmaxf(v, dpp_rot<0x121>(v));   // row_ror:1
  return v;
}
__device__ __forceinline__ float rowsum16(float v) {
  v += dpp_rot<0x128>(v);
  v += dpp_rot<0x124>(v);
  v += dpp_rot<0x122>(v);
  v += dpp_rot<0x121>(v);
  return v;
}

// ================================================================ generic GEMM
// C[M,N] = A[M,K] @ W[N,K]^T (+bias) ; f32 in/out, f16 WMMA with f32 accum.
// Block tile 128x128, BK=32, 8 waves (4x2), each wave 32x64 = 2x4 wmma tiles.
// Register-staged double buffering: global loads for tile k+1 issue while
// WMMA consumes tile k from LDS.
// MODE 0: row-major store C[m*N+n]. MODE 1: value-permute store for the
//         deformable value tensor: (b*STOT+s, h*32+d) -> value[b][h][s][d].
#define GP 40   // LDS pitch in halves (32 + 8 pad, keeps 16B alignment: 80B rows)

template<int MODE, int RELU>
__launch_bounds__(256, 2)
__global__ void gemm_f16wmma(const float* __restrict__ A, const float* __restrict__ W,
                             const float* __restrict__ bias, float* __restrict__ C,
                             int M, int N, int K) {
  __shared__ alignas(16) _Float16 As[128 * GP];
  __shared__ alignas(16) _Float16 Bs[128 * GP];
  const int tid  = threadIdx.x;
  const int nT   = N >> 7;
  const int bm0  = (blockIdx.x / nT) << 7;
  const int bn0  = (blockIdx.x % nT) << 7;
  const int wv   = tid >> 5, lane = tid & 31;
  const int mg   = wv >> 1,  ng   = wv & 1;
  const int hf   = lane >> 4, l16 = lane & 15;

  v8f acc[2][4];
#pragma unroll
  for (int a = 0; a < 2; ++a)
#pragma unroll
    for (int b = 0; b < 4; ++b)
#pragma unroll
      for (int i = 0; i < 8; ++i) acc[a][b][i] = 0.f;

  float4 avr[4], bvr[4];
  auto stage = [&](int k0) {
#pragma unroll
    for (int it = 0; it < 4; ++it) {
      int slot = it * 256 + tid;          // 0..1023: 128 rows x 8 float4
      int r    = slot >> 3;
      int c4   = (slot & 7) << 2;
      avr[it] = *(const float4*)(A + (size_t)(bm0 + r) * K + k0 + c4);
      bvr[it] = *(const float4*)(W + (size_t)(bn0 + r) * K + k0 + c4);
    }
  };
  stage(0);

  for (int k0 = 0; k0 < K; k0 += 32) {
    __syncthreads();   // previous tile fully consumed
    // commit staged registers to LDS with f32 -> f16 conversion
#pragma unroll
    for (int it = 0; it < 4; ++it) {
      int slot = it * 256 + tid;
      int r    = slot >> 3;
      int c4   = (slot & 7) << 2;
      _Float16* da = &As[r * GP + c4];
      da[0] = (_Float16)avr[it].x; da[1] = (_Float16)avr[it].y;
      da[2] = (_Float16)avr[it].z; da[3] = (_Float16)avr[it].w;
      _Float16* db = &Bs[r * GP + c4];
      db[0] = (_Float16)bvr[it].x; db[1] = (_Float16)bvr[it].y;
      db[2] = (_Float16)bvr[it].z; db[3] = (_Float16)bvr[it].w;
    }
    __syncthreads();
    if (k0 + 32 < K) stage(k0 + 32);   // prefetch next tile under the compute

    Frag16 af[2], bf[4];
#pragma unroll
    for (int mt = 0; mt < 2; ++mt) {   // A frag: k = {hf*8..+7} u {16+hf*8..+7}
      const _Float16* s = &As[(mg * 32 + mt * 16 + l16) * GP + hf * 8];
      af[mt].h[0] = *(const v8h*)s;
      af[mt].h[1] = *(const v8h*)(s + 16);
    }
#pragma unroll
    for (int nt = 0; nt < 4; ++nt) {   // B frag: k = {hf*8..+7} u {16+hf*8..+7}
      const _Float16* s = &Bs[(ng * 64 + nt * 16 + l16) * GP + hf * 8];
      bf[nt].h[0] = *(const v8h*)s;
      bf[nt].h[1] = *(const v8h*)(s + 16);
    }
#pragma unroll
    for (int mt = 0; mt < 2; ++mt)
#pragma unroll
      for (int nt = 0; nt < 4; ++nt)
        acc[mt][nt] = __builtin_amdgcn_wmma_f32_16x16x32_f16(
            false, af[mt].v, false, bf[nt].v, (short)0, acc[mt][nt], false, false);
  }

  // epilogue (C layout: m = i + 8*hf, n = l16 within a 16x16 tile)
#pragma unroll
  for (int nt = 0; nt < 4; ++nt) {
    int gn = bn0 + ng * 64 + nt * 16 + l16;
    float bvad = bias ? bias[gn] : 0.f;
#pragma unroll
    for (int mt = 0; mt < 2; ++mt) {
#pragma unroll
      for (int i = 0; i < 8; ++i) {
        int gm = bm0 + mg * 32 + mt * 16 + i + hf * 8;
        float val = acc[mt][nt][i] + bvad;
        if (RELU) val = fmaxf(val, 0.f);
        if (MODE == 0) {
          C[(size_t)gm * N + gn] = val;
        } else {
          int bb = gm / STOT;
          int s  = gm - bb * STOT;
          int hh = gn >> 5, d = gn & 31;
          C[(((size_t)bb * NHc + hh) * STOT + s) * DHc + d] = val;
        }
      }
    }
  }
}

// ================================================================ flash attention
// grid: B*NH*(Q/128); 8 waves, each owns 16 complete query rows -> online
// softmax state never leaves the wave (DPP row reductions, VALU only).
__launch_bounds__(256, 2)
__global__ void attn_kernel(const float* __restrict__ q, const float* __restrict__ k,
                            const float* __restrict__ v, float* __restrict__ ctx) {
  __shared__ alignas(16) _Float16 Vt[32 * 72];    // V chunk transposed: [d][kk]
  __shared__ alignas(16) _Float16 Pl[128 * 72];   // probabilities, wave-private rows
  const int tid = threadIdx.x;
  int bx = blockIdx.x;
  const int qt = bx & 7;
  const int hh = (bx >> 3) & 7;
  const int b  = bx >> 6;
  const int wv = tid >> 5, lane = tid & 31;
  const int hf = lane >> 4, l16 = lane & 15;
  const int q0 = qt * 128 + wv * 16;

  Frag16 aq;   // query fragment (rows fixed for whole kernel)
  {
    const float* qr = q + ((size_t)(b * Qc + q0 + l16)) * Ec + hh * DHc;
#pragma unroll
    for (int j = 0; j < 8; ++j) {
      int kk = ((j >> 2) << 4) + hf * 8 + ((j & 3) << 1);
      aq.e[2 * j]     = (_Float16)qr[kk];
      aq.e[2 * j + 1] = (_Float16)qr[kk + 1];
    }
  }

  float mi[8], li[8];
  v8f o0, o1, zacc;
#pragma unroll
  for (int i = 0; i < 8; ++i) {
    mi[i] = -1e30f; li[i] = 0.f; o0[i] = 0.f; o1[i] = 0.f; zacc[i] = 0.f;
  }
  const float scale = 0.1767766952966369f;   // 1/sqrt(32)

  for (int kk0 = 0; kk0 < Qc; kk0 += 64) {
    __syncthreads();
    // stage V chunk transposed (coalesced global reads across d)
#pragma unroll
    for (int e2 = 0; e2 < 8; ++e2) {
      int lin = e2 * 256 + tid;
      int r = lin >> 5, d = lin & 31;
      Vt[d * 72 + r] =
          (_Float16)v[((size_t)(b * Qc + kk0 + r)) * Ec + hh * DHc + d];
    }
    __syncthreads();

    // S = q @ K^T : 4 column tiles of 16 keys
    v8f s[4];
#pragma unroll
    for (int j = 0; j < 4; ++j) {
      Frag16 bk;   // B frag: lane owns key row, k = {hf*8..+7} u {16+hf*8..+7} over DH
      const float* kr = k + ((size_t)(b * Qc + kk0 + j * 16 + l16)) * Ec + hh * DHc;
#pragma unroll
      for (int i = 0; i < 16; ++i) bk.e[i] = (_Float16)kr[((i >> 3) << 4) + hf * 8 + (i & 7)];
      s[j] = __builtin_amdgcn_wmma_f32_16x16x32_f16(
          false, aq.v, false, bk.v, (short)0, zacc, false, false);
    }

    // online softmax (per output row = per (i, lane-half) slot) - DPP only
#pragma unroll
    for (int i = 0; i < 8; ++i) {
#pragma unroll
      for (int j = 0; j < 4; ++j) s[j][i] *= scale;
      float rm = fmaxf(fmaxf(s[0][i], s[1][i]), fmaxf(s[2][i], s[3][i]));
      rm = rowmax16(rm);
      float mn = fmaxf(mi[i], rm);
      float sf = __expf(mi[i] - mn);
      mi[i] = mn;
      float rs = 0.f;
#pragma unroll
      for (int j = 0; j < 4; ++j) {
        float p = __expf(s[j][i] - mn);
        s[j][i] = p;
        rs += p;
      }
      rs = rowsum16(rs);
      li[i] = li[i] * sf + rs;
      o0[i] *= sf;
      o1[i] *= sf;
    }

    // re-layout P (C layout) -> A layout via wave-private LDS rows
#pragma unroll
    for (int i = 0; i < 8; ++i)
#pragma unroll
      for (int j = 0; j < 4; ++j)
        Pl[(wv * 16 + i + hf * 8) * 72 + j * 16 + l16] = (_Float16)s[j][i];
    asm volatile("s_wait_dscnt 0x0" ::: "memory");

    // O += P @ V (K = 64 -> two K=32 wmma steps, two d-tiles)
#pragma unroll
    for (int ko = 0; ko < 64; ko += 32) {
      Frag16 ap, b0, b1;
      const _Float16* ps = &Pl[(wv * 16 + l16) * 72 + ko + hf * 8];
      ap.h[0] = *(const v8h*)ps;
      ap.h[1] = *(const v8h*)(ps + 16);
      const _Float16* v0 = &Vt[l16 * 72 + ko + hf * 8];
      b0.h[0] = *(const v8h*)v0;
      b0.h[1] = *(const v8h*)(v0 + 16);
      const _Float16* v1 = &Vt[(16 + l16) * 72 + ko + hf * 8];
      b1.h[0] = *(const v8h*)v1;
      b1.h[1] = *(const v8h*)(v1 + 16);
      o0 = __builtin_amdgcn_wmma_f32_16x16x32_f16(false, ap.v, false, b0.v, (short)0, o0, false, false);
      o1 = __builtin_amdgcn_wmma_f32_16x16x32_f16(false, ap.v, false, b1.v, (short)0, o1, false, false);
    }
  }

#pragma unroll
  for (int i = 0; i < 8; ++i) {
    float inv = 1.f / li[i];
    size_t base = ((size_t)(b * Qc + q0 + i + hf * 8)) * Ec + hh * DHc;
    ctx[base + l16]      = o0[i] * inv;
    ctx[base + 16 + l16] = o1[i] * inv;
  }
}

// ================================================================ elementwise / LN
__global__ void add_kernel(const float* __restrict__ a, const float* __restrict__ b,
                           float* __restrict__ o) {
  size_t i = (size_t)blockIdx.x * blockDim.x + threadIdx.x;
  o[i] = a[i] + b[i];
}

// out = LN(a + r) * g + be ; optional out2 = out + qp (query = tgt + query_pos)
__launch_bounds__(256)
__global__ void ln_kernel(const float* __restrict__ a, const float* __restrict__ r,
                          const float* __restrict__ g, const float* __restrict__ be,
                          float* __restrict__ out, const float* __restrict__ qp,
                          float* __restrict__ out2) {
  __shared__ float red[16];
  const int tid = threadIdx.x;
  const size_t idx = (size_t)blockIdx.x * 256 + tid;
  float x = a[idx] + r[idx];
  float s  = rowsum16(x);            // DPP within 16-lane rows
  float s2 = rowsum16(x * x);
  s  += __shfl_xor(s, 16);           // single cross-row exchange
  s2 += __shfl_xor(s2, 16);
  if ((tid & 31) == 0) { red[(tid >> 5) * 2] = s; red[(tid >> 5) * 2 + 1] = s2; }
  __syncthreads();
  float ts = 0.f, ts2 = 0.f;
#pragma unroll
  for (int i = 0; i < 8; ++i) { ts += red[2 * i]; ts2 += red[2 * i + 1]; }
  float mu  = ts * (1.f / 256.f);
  float var = ts2 * (1.f / 256.f) - mu * mu;
  float rsd = rsqrtf(var + 1e-5f);
  float y = (x - mu) * rsd * g[tid] + be[tid];
  out[idx] = y;
  if (out2) out2[idx] = y + qp[idx];
}

// ================================================================ aw softmax + locations
// one thread per (b,q,h): softmax over L*P=16 logits; precompute sample locs.
__global__ void awloc_kernel(const float* __restrict__ awl, const float* __restrict__ off,
                             const float* __restrict__ rp, float* __restrict__ awsm,
                             float* __restrict__ loc) {
  int t = blockIdx.x * 256 + threadIdx.x;    // B*Q*NH = 65536
  int hh = t & 7, qi = (t >> 3) & 1023, b = t >> 13;
  const float* lg = awl + ((size_t)(b * Qc + qi)) * 128 + hh * 16;
  float e[16];
  float mx = lg[0];
#pragma unroll
  for (int i = 1; i < 16; ++i) mx = fmaxf(mx, lg[i]);
  float sum = 0.f;
#pragma unroll
  for (int i = 0; i < 16; ++i) { e[i] = __expf(lg[i] - mx); sum += e[i]; }
  float inv = 1.f / sum;
  float* ao = awsm + (((size_t)(b * NHc + hh)) * Qc + qi) * 16;
#pragma unroll
  for (int i = 0; i < 16; ++i) ao[i] = e[i] * inv;

  const float* ob = off + ((size_t)(b * Qc + qi)) * 256 + hh * 32;
  const float* rb = rp  + ((size_t)(b * Qc + qi)) * 8;
  float* lo = loc + (((size_t)(b * NHc + hh)) * Qc + qi) * 32;
#pragma unroll
  for (int l = 0; l < 4; ++l) {
    float wn = (float)d_HL[l];                // square levels: W == H
    float rx = rb[l * 2], ry = rb[l * 2 + 1];
#pragma unroll
    for (int p = 0; p < 4; ++p) {
      int lp = l * 4 + p;
      lo[lp * 2]     = rx + ob[lp * 2]     / wn;
      lo[lp * 2 + 1] = ry + ob[lp * 2 + 1] / wn;
    }
  }
}

// ================================================================ bilinear sampling
__device__ __forceinline__ float fetchv(const float* vb, int s0, int Wi,
                                        int yi, int xi, int d) {
  bool ok = (xi >= 0) && (xi < Wi) && (yi >= 0) && (yi < Wi);
  int xc = min(max(xi, 0), Wi - 1);
  int yc = min(max(yi, 0), Wi - 1);
  float val = vb[((size_t)(s0 + yc * Wi + xc)) * DHc + d];
  return ok ? val : 0.f;
}

// one thread per (b,h,q,d): 32 consecutive d-lanes hit 128B-contiguous value rows
__global__ void sample_kernel(const float* __restrict__ val, const float* __restrict__ aw,
                              const float* __restrict__ loc, float* __restrict__ cain) {
  int t  = blockIdx.x * 256 + threadIdx.x;   // 2M threads
  int d  = t & 31, qi = (t >> 5) & 1023, hh = (t >> 15) & 7, b = t >> 18;
  const float* ab = aw  + (((size_t)(b * NHc + hh)) * Qc + qi) * 16;
  const float* lb = loc + (((size_t)(b * NHc + hh)) * Qc + qi) * 32;
  const float* vb = val + ((size_t)(b * NHc + hh)) * STOT * DHc;
  float acc = 0.f;
#pragma unroll
  for (int lp = 0; lp < 16; ++lp) {
    int l = lp >> 2;
    int Wi = d_HL[l], s0 = d_ST[l];
    float Wf = (float)Wi;
    float x = lb[lp * 2] * Wf - 0.5f;
    float y = lb[lp * 2 + 1] * Wf - 0.5f;
    float x0f = floorf(x), y0f = floorf(y);
    float wx = x - x0f, wy = y - y0f;
    int x0 = (int)x0f, y0 = (int)y0f;
    float v00 = fetchv(vb, s0, Wi, y0,     x0,     d);
    float v01 = fetchv(vb, s0, Wi, y0,     x0 + 1, d);
    float v10 = fetchv(vb, s0, Wi, y0 + 1, x0,     d);
    float v11 = fetchv(vb, s0, Wi, y0 + 1, x0 + 1, d);
    float bil = v00 * (1.f - wy) * (1.f - wx) + v01 * (1.f - wy) * wx
              + v10 * wy * (1.f - wx)         + v11 * wy * wx;
    acc += ab[lp] * bil;
  }
  cain[((size_t)(b * Qc + qi)) * Ec + hh * DHc + d] = acc;
}

// ================================================================ launch
extern "C" void kernel_launch(void* const* d_in, const int* in_sizes, int n_in,
                              void* d_out, int out_size, void* d_ws, size_t ws_size,
                              hipStream_t stream) {
  (void)in_sizes; (void)n_in; (void)out_size; (void)ws_size;
  const float* tgt   = (const float*)d_in[0];
  const float* qp    = (const float*)d_in[1];
  const float* rp    = (const float*)d_in[2];
  const float* src   = (const float*)d_in[3];
  // d_in[4] spatial_shapes, d_in[5] level_start_index: compile-time constants
  const float* ipw   = (const float*)d_in[6];
  const float* ipb   = (const float*)d_in[7];
  const float* opw   = (const float*)d_in[8];
  const float* opb   = (const float*)d_in[9];
  const float* n1g   = (const float*)d_in[10];
  const float* n1b   = (const float*)d_in[11];
  const float* offw  = (const float*)d_in[12];
  const float* offb_ = (const float*)d_in[13];
  const float* aww   = (const float*)d_in[14];
  const float* awb   = (const float*)d_in[15];
  const float* vpw   = (const float*)d_in[16];
  const float* vpb   = (const float*)d_in[17];
  const float* opw2  = (const float*)d_in[18];
  const float* opb2  = (const float*)d_in[19];
  const float* n2g   = (const float*)d_in[20];
  const float* n2b   = (const float*)d_in[21];
  const float* w1    = (const float*)d_in[22];
  const float* b1    = (const float*)d_in[23];
  const float* w2    = (const float*)d_in[24];
  const float* b2    = (const float*)d_in[25];
  const float* n3g   = (const float*)d_in[26];
  const float* n3b   = (const float*)d_in[27];

  float* ws = (float*)d_ws;
  const size_t M1 = (size_t)1 << 20;
  // buffer reuse plan (each slot 2M floats unless noted)
  float* qk    = ws;            float* ctxb = qk;        float* cain = qk;
  float* qb    = ws + 2 * M1;   float* offo = qb;        float* y3   = qb;
  float* kb    = ws + 4 * M1;   float* awl  = kb;        float* awsm = kb + M1;
  float* vb    = ws + 6 * M1;   float* locb = vb;
  float* y1    = ws + 8 * M1;   float* y2   = y1;
  float* tgt1  = ws + 10 * M1;
  float* query = ws + 12 * M1;
  float* tgt2  = ws + 14 * M1;
  float* ffh   = ws + 16 * M1;              // 8M floats
  float* valb  = ws + 24 * M1;              // 44.56M floats (B,NH,S,DH)

  dim3 blk(256);

  // qk = tgt + query_pos
  add_kernel<<<8192, blk, 0, stream>>>(tgt, qp, qk);

  // q/k/v projections
  gemm_f16wmma<0,0><<<128, blk, 0, stream>>>(qk,  ipw,           ipb,       qb, 8192, 256, 256);
  gemm_f16wmma<0,0><<<128, blk, 0, stream>>>(qk,  ipw +  65536,  ipb + 256, kb, 8192, 256, 256);
  gemm_f16wmma<0,0><<<128, blk, 0, stream>>>(tgt, ipw + 131072,  ipb + 512, vb, 8192, 256, 256);

  // fused self-attention (flash, online softmax)
  attn_kernel<<<512, blk, 0, stream>>>(qb, kb, vb, ctxb);

  // out_proj + residual LN1 (also emits query = tgt1 + query_pos)
  gemm_f16wmma<0,0><<<128, blk, 0, stream>>>(ctxb, opw, opb, y1, 8192, 256, 256);
  ln_kernel<<<8192, blk, 0, stream>>>(tgt, y1, n1g, n1b, tgt1, qp, query);

  // value projection with (b,h,s,d) permuted store (the big GEMM, 22.8 GFLOP)
  gemm_f16wmma<1,0><<<2720, blk, 0, stream>>>(src, vpw, vpb, valb, Bc * STOT, 256, 256);

  // offsets / attention-weight logits
  gemm_f16wmma<0,0><<<128, blk, 0, stream>>>(query, offw, offb_, offo, 8192, 256, 256);
  gemm_f16wmma<0,0><<< 64, blk, 0, stream>>>(query, aww,  awb,   awl,  8192, 128, 256);

  // softmax(16) + sampling locations; then bilinear gather + weighted sum
  awloc_kernel<<<256, blk, 0, stream>>>(awl, offo, rp, awsm, locb);
  sample_kernel<<<8192, blk, 0, stream>>>(valb, awsm, locb, cain);

  // output projection + residual LN2
  gemm_f16wmma<0,0><<<128, blk, 0, stream>>>(cain, opw2, opb2, y2, 8192, 256, 256);
  ln_kernel<<<8192, blk, 0, stream>>>(tgt1, y2, n2g, n2b, tgt2, nullptr, nullptr);

  // FFN + residual LN3 -> d_out
  gemm_f16wmma<0,1><<<512, blk, 0, stream>>>(tgt2, w1, b1, ffh, 8192, 1024, 256);
  gemm_f16wmma<0,0><<<128, blk, 0, stream>>>(ffh,  w2, b2, y3,  8192, 256, 1024);
  ln_kernel<<<8192, blk, 0, stream>>>(tgt2, y3, n3g, n3b, (float*)d_out, nullptr, nullptr);
}
